// MultiHeadAttention_59837484368468
// MI455X (gfx1250) — matrix operand order repaired and checked by reading, not compiled, not run on hardware
//
#include <hip/hip_runtime.h>
#include <stdint.h>

// ---------- types ----------
typedef __attribute__((ext_vector_type(16))) __bf16        bf16x16;
typedef __attribute__((ext_vector_type(8)))  float         f32x8;
typedef __attribute__((ext_vector_type(4)))  unsigned int  u32x4;
typedef __attribute__((ext_vector_type(2)))  unsigned int  u32x2;
typedef __attribute__((ext_vector_type(4)))  int           i32x4;
typedef __attribute__((ext_vector_type(8)))  int           i32x8;
typedef unsigned short bf16u;   // storage type for bf16 (host-safe)

struct U32x8 { u32x4 lo, hi; };

static __device__ __forceinline__ unsigned short f2bfu(float f) {
  unsigned u = __builtin_bit_cast(unsigned, f);
  u = (u + 0x7FFFu + ((u >> 16) & 1u)) >> 16;
  return (unsigned short)u;
}

static __device__ __forceinline__ u32x4 ld4(const bf16u* p) { return *(const u32x4*)p; }
static __device__ __forceinline__ void  st4(bf16u* p, u32x4 v) { *(u32x4*)p = v; }

static __device__ __forceinline__ bf16x16 join16(u32x4 lo, u32x4 hi) {
  U32x8 t; t.lo = lo; t.hi = hi;
  return __builtin_bit_cast(bf16x16, t);
}

static __device__ __forceinline__ f32x8 zero8() {
  f32x8 z = {0.f, 0.f, 0.f, 0.f, 0.f, 0.f, 0.f, 0.f};
  return z;
}

static __device__ __forceinline__ f32x8 wmma_bf16(bf16x16 a, bf16x16 b, f32x8 c) {
  // D = A(16x32 bf16) x B(32x16 bf16) + C(16x16 f32)
  return __builtin_amdgcn_wmma_f32_16x16x32_bf16(
      /*neg_a=*/false, a, /*neg_b=*/false, b,
      /*c_mod=*/(short)0, c, /*reuse_a=*/false, /*reuse_b=*/false);
}

// ---------- TDM: 2D bf16 tile load, global -> LDS (one issue per wave) ----------
// D# group0: [1:0]=count=1, [63:32]=lds_addr, [120:64]=global_addr, [127:126]=type=2
// D# group1: data_size=1(2B), tensor_dim0/1 = tile dims (tile fully in-bounds),
//            tile_dim0/1, tensor_dim0_stride = row pitch (elements).
static __device__ __forceinline__ void tdm_load_2d_bf16(
    const bf16u* gsrc, bf16u* lds, int tile_w, int tile_h, int pitch_elems) {
  size_t ga = (size_t)gsrc;
  i32x4 g0;
  g0[0] = 1;                                            // count=1
  g0[1] = (int)(unsigned)(size_t)lds;                   // lds byte offset
  g0[2] = (int)(unsigned)(ga & 0xFFFFFFFFu);            // global_addr[31:0]
  g0[3] = (int)((unsigned)((ga >> 32) & 0x1FFFFFFu) | 0x80000000u);  // [56:32] | type=2
  i32x8 g1;
  g1[0] = 0x00010000;                                   // data_size=1 -> 2 bytes
  g1[1] = (tile_w & 0xFFFF) << 16;                      // tensor_dim0[15:0]
  g1[2] = ((tile_w >> 16) & 0xFFFF) | ((tile_h & 0xFFFF) << 16);  // dim0 hi | dim1 lo
  g1[3] = ((tile_h >> 16) & 0xFFFF) | ((tile_w & 0xFFFF) << 16);  // dim1 hi | tile_dim0
  g1[4] = (tile_h & 0xFFFF);                            // tile_dim1, tile_dim2=0
  g1[5] = pitch_elems;                                  // tensor_dim0_stride[31:0]
  g1[6] = 0;                                            // stride hi, dim1_stride lo
  g1[7] = 0;
  i32x4 gz = {0, 0, 0, 0};
  asm volatile("tensor_load_to_lds %0, %1, %2, %3"
               :
               : "s"(g0), "s"(g1), "s"(gz), "s"(gz)
               : "memory");
}

static __device__ __forceinline__ void wait_tensorcnt0() {
#if __has_builtin(__builtin_amdgcn_s_wait_tensorcnt)
  __builtin_amdgcn_s_wait_tensorcnt(0);
#else
  asm volatile("s_wait_tensorcnt 0x0" ::: "memory");
#endif
}

// ---------- fp32 -> bf16 conversion (4 elements / thread) ----------
__global__ __launch_bounds__(256) void mha_cvt4_kernel(
    const float* __restrict__ src, bf16u* __restrict__ dst, int n4) {
  int i = blockIdx.x * blockDim.x + threadIdx.x;
  if (i < n4) {
    const float4 f = ((const float4*)src)[i];
    u32x2 o;
    o.x = (unsigned)f2bfu(f.x) | ((unsigned)f2bfu(f.y) << 16);
    o.y = (unsigned)f2bfu(f.z) | ((unsigned)f2bfu(f.w) << 16);
    ((u32x2*)dst)[i] = o;
  }
}

// ---------- bf16 GEMM: C[MxN] = A[MxK] * Bw[KxN], fp32 accumulate ----------
// Workgroup = 256 threads (8 waves), tile 128(M) x 128(N), K slab = 32.
// A slab loaded by TDM (wave 0); B slab loaded+transposed by all threads.
template <bool OUT_BF16>
__global__ __launch_bounds__(256) void mha_gemm_bf16_kernel(
    const bf16u* __restrict__ A, const bf16u* __restrict__ Bw,
    void* __restrict__ Cout, int M, int N, int K) {
  __shared__ bf16u sA[128 * 32];  // [m][k]
  __shared__ bf16u sB[128 * 32];  // [n][k] (transposed on fill)

  const int tid   = threadIdx.x;
  const int wave  = tid >> 5, lane = tid & 31;
  const int wy    = wave >> 1, wx = wave & 1;
  const int r     = lane & 15;
  const int abase = (lane < 16) ? 0 : 8;    // A-frag K-half (interleaved 8s)
  const int bbase = (lane < 16) ? 0 : 8;    // B-frag K-half (interleaved 8s)
  const int m0 = blockIdx.y * 128, n0 = blockIdx.x * 128;

  f32x8 acc[2][4];
#pragma unroll
  for (int i = 0; i < 2; ++i)
#pragma unroll
    for (int j = 0; j < 4; ++j) acc[i][j] = zero8();

  for (int k0 = 0; k0 < K; k0 += 32) {
    __syncthreads();
    // A tile 128x32 via Tensor Data Mover (one descriptor, wave 0 issues)
    if (wave == 0)
      tdm_load_2d_bf16(A + (size_t)m0 * K + k0, sA, /*w=*/32, /*h=*/128, /*pitch=*/K);
    // B tile: read [k][n] coalesced, store transposed as [n][k]
#pragma unroll
    for (int i = 0; i < 2; ++i) {
      int c = tid + i * 256;
      int kr = c >> 4, nc = c & 15;
      u32x4 v = ld4(Bw + (size_t)(k0 + kr) * N + n0 + nc * 8);
      const bf16u* p = (const bf16u*)&v;
#pragma unroll
      for (int j = 0; j < 8; ++j) sB[(nc * 8 + j) * 32 + kr] = p[j];
      if (k0 + 32 < K)  // prefetch next weight slab -> global_prefetch_b8
        __builtin_prefetch(Bw + (size_t)(k0 + 32 + kr) * N + n0 + nc * 8, 0, 3);
    }
    if (wave == 0) wait_tensorcnt0();
    __syncthreads();

    bf16x16 af[2], bfr[4];
#pragma unroll
    for (int mt = 0; mt < 2; ++mt) {
      const bf16u* p = sA + (wy * 32 + mt * 16 + r) * 32 + abase;
      af[mt] = join16(ld4(p), ld4(p + 16));
    }
#pragma unroll
    for (int nt = 0; nt < 4; ++nt) {
      const bf16u* p = sB + (wx * 64 + nt * 16 + r) * 32 + bbase;
      bfr[nt] = join16(ld4(p), ld4(p + 16));
    }
#pragma unroll
    for (int mt = 0; mt < 2; ++mt)
#pragma unroll
      for (int nt = 0; nt < 4; ++nt)
        acc[mt][nt] = wmma_bf16(af[mt], bfr[nt], acc[mt][nt]);
  }

  // epilogue (C layout: row = e + abase, col = lane&15)
  const int mr0 = m0 + wy * 32, nc0 = n0 + wx * 64;
#pragma unroll
  for (int mt = 0; mt < 2; ++mt)
#pragma unroll
    for (int nt = 0; nt < 4; ++nt)
#pragma unroll
      for (int e = 0; e < 8; ++e) {
        size_t idx = (size_t)(mr0 + mt * 16 + abase + e) * N + (nc0 + nt * 16 + r);
        if (OUT_BF16) ((bf16u*)Cout)[idx] = f2bfu(acc[mt][nt][e]);
        else          ((float*)Cout)[idx] = acc[mt][nt][e];
      }
}

// ---------- causal flash attention ----------
// grid = (S/64, H, B); 128 threads = 4 waves; wave w owns query rows [qbase+16w, +16)
__global__ __launch_bounds__(128) void mha_attn_kernel(
    const bf16u* __restrict__ Qb, const bf16u* __restrict__ Kb,
    const bf16u* __restrict__ Vb, bf16u* __restrict__ ctx,
    int S, int D, int H) {
  constexpr int DH = 64;
  __shared__ bf16u sK[64 * DH];       // [key][dh]  (TDM-filled)
  __shared__ bf16u sVt[DH * 64];      // [dh][key]  (transposed on fill)
  __shared__ bf16u sP[4][16 * 64];    // per-wave P staging [q][key]

  const int tid   = threadIdx.x;
  const int wave  = tid >> 5, lane = tid & 31;
  const int r     = lane & 15;
  const int abase = (lane < 16) ? 0 : 8;
  const int bbase = (lane < 16) ? 0 : 8;
  const int qt = blockIdx.x, h = blockIdx.y, b = blockIdx.z;
  const int qrow = qt * 64 + wave * 16;

  // Q fragments for dh-halves 0..31 and 32..63
  bf16x16 qf[2];
  {
    const bf16u* qp = Qb + (size_t)(b * S + qrow + r) * D + h * DH;
#pragma unroll
    for (int kd = 0; kd < 2; ++kd) {
      const bf16u* p = qp + kd * 32 + abase;
      qf[kd] = join16(ld4(p), ld4(p + 16));
    }
  }

  float m_st[8], l_st[8];
  f32x8 o[4];
#pragma unroll
  for (int e = 0; e < 8; ++e) { m_st[e] = -__builtin_inff(); l_st[e] = 0.f; }
#pragma unroll
  for (int n = 0; n < 4; ++n) o[n] = zero8();

  for (int kt = 0; kt <= qt; ++kt) {
    __syncthreads();
    // K tile 64x64 via TDM (wave 0); V tile transposed manually by all threads
    if (wave == 0)
      tdm_load_2d_bf16(Kb + (size_t)(b * S + kt * 64) * D + h * DH, sK,
                       /*w=*/DH, /*h=*/64, /*pitch=*/D);
#pragma unroll
    for (int i = 0; i < 4; ++i) {
      int c = tid + i * 128;                 // 512 chunks of 8 bf16
      int row = c >> 3, cc = c & 7;
      size_t g = (size_t)(b * S + kt * 64 + row) * D + h * DH + cc * 8;
      u32x4 vv = ld4(Vb + g);
      const bf16u* p = (const bf16u*)&vv;
#pragma unroll
      for (int j = 0; j < 8; ++j) sVt[(cc * 8 + j) * 64 + row] = p[j];
    }
    if (wave == 0) wait_tensorcnt0();
    __syncthreads();

    // scores S = Q * K^T  (16 queries x 64 keys per wave)
    f32x8 sacc[4];
#pragma unroll
    for (int n = 0; n < 4; ++n) {
      sacc[n] = zero8();
#pragma unroll
      for (int kd = 0; kd < 2; ++kd) {
        const bf16u* p = sK + (n * 16 + r) * 64 + kd * 32 + bbase;
        sacc[n] = wmma_bf16(qf[kd], join16(ld4(p), ld4(p + 16)), sacc[n]);
      }
    }

    // scale by 1/sqrt(64), then mask (diagonal tile only) to -inf
    const bool diag = (kt == qt);
#pragma unroll
    for (int n = 0; n < 4; ++n)
#pragma unroll
      for (int e = 0; e < 8; ++e) {
        float v = sacc[n][e] * 0.125f;
        if (diag && (kt * 64 + n * 16 + r) > (qrow + abase + e)) v = -__builtin_inff();
        sacc[n][e] = v;
      }

    // online softmax: row lives in element e across 16 lanes of a half-wave
    float alpha[8];
#pragma unroll
    for (int e = 0; e < 8; ++e) {
      float v = fmaxf(fmaxf(sacc[0][e], sacc[1][e]), fmaxf(sacc[2][e], sacc[3][e]));
#pragma unroll
      for (int off = 1; off < 16; off <<= 1) v = fmaxf(v, __shfl_xor(v, off, 32));
      float nm = fmaxf(m_st[e], v);
      alpha[e] = __expf(m_st[e] - nm);
      m_st[e] = nm;
    }
    float rs[8];
#pragma unroll
    for (int e = 0; e < 8; ++e) rs[e] = 0.f;
#pragma unroll
    for (int n = 0; n < 4; ++n)
#pragma unroll
      for (int e = 0; e < 8; ++e) {
        float p = __expf(sacc[n][e] - m_st[e]);
        sacc[n][e] = p;
        rs[e] += p;
      }
#pragma unroll
    for (int e = 0; e < 8; ++e) {
      float v = rs[e];
#pragma unroll
      for (int off = 1; off < 16; off <<= 1) v += __shfl_xor(v, off, 32);
      l_st[e] = l_st[e] * alpha[e] + v;
    }
#pragma unroll
    for (int n = 0; n < 4; ++n)
#pragma unroll
      for (int e = 0; e < 8; ++e) o[n][e] *= alpha[e];

    // C-layout -> A-layout via per-wave LDS staging
    bf16u* pw = sP[wave];
#pragma unroll
    for (int n = 0; n < 4; ++n)
#pragma unroll
      for (int e = 0; e < 8; ++e)
        pw[(abase + e) * 64 + n * 16 + r] = f2bfu(sacc[n][e]);
    asm volatile("s_wait_dscnt 0" ::: "memory");

    // O += P * V
#pragma unroll
    for (int kd = 0; kd < 2; ++kd) {
      const bf16u* pa = pw + r * 64 + kd * 32 + abase;
      bf16x16 pf = join16(ld4(pa), ld4(pa + 16));
#pragma unroll
      for (int n = 0; n < 4; ++n) {
        const bf16u* pv = sVt + (n * 16 + r) * 64 + kd * 32 + bbase;
        o[n] = wmma_bf16(pf, join16(ld4(pv), ld4(pv + 16)), o[n]);
      }
    }
  }

  // normalize, stage to LDS, coalesced store as [B][H][S][DH]
  // (flat-identical to the reference's head-mixing reshape to [B,S,D])
  bf16u* pw = sP[wave];
#pragma unroll
  for (int n = 0; n < 4; ++n)
#pragma unroll
    for (int e = 0; e < 8; ++e)
      pw[(abase + e) * 64 + n * 16 + r] = f2bfu(o[n][e] / l_st[e]);
  asm volatile("s_wait_dscnt 0" ::: "memory");

  bf16u* dst = ctx + ((size_t)(b * H + h) * S + qrow) * DH;
#pragma unroll
  for (int i = 0; i < 4; ++i) {
    int idx = lane + i * 32;  // 128 chunks of 8 bf16 = contiguous 2 KB block
    st4(dst + idx * 8, ld4(pw + idx * 8));
  }
}

// ---------- host launcher ----------
extern "C" void kernel_launch(void* const* d_in, const int* in_sizes, int n_in,
                              void* d_out, int out_size, void* d_ws, size_t ws_size,
                              hipStream_t stream) {
  (void)in_sizes; (void)n_in; (void)out_size; (void)ws_size;
  const int B = 2, S = 2048, D = 1024, H = 16;
  const int M = B * S;

  const float* x  = (const float*)d_in[0];
  const float* wq = (const float*)d_in[1];
  const float* wk = (const float*)d_in[2];
  const float* wv = (const float*)d_in[3];
  const float* wo = (const float*)d_in[4];

  char* ws = (char*)d_ws;
  size_t off = 0;
  auto alloc = [&](size_t bytes) {
    char* p = ws + off;
    off += (bytes + 255) & ~(size_t)255;
    return p;
  };
  bf16u* xb  = (bf16u*)alloc((size_t)M * D * 2);
  bf16u* wqb = (bf16u*)alloc((size_t)D * D * 2);
  bf16u* wkb = (bf16u*)alloc((size_t)D * D * 2);
  bf16u* wvb = (bf16u*)alloc((size_t)D * D * 2);
  bf16u* wob = (bf16u*)alloc((size_t)D * D * 2);
  bf16u* Qb  = (bf16u*)alloc((size_t)M * D * 2);
  bf16u* Kb  = (bf16u*)alloc((size_t)M * D * 2);
  bf16u* Vb  = (bf16u*)alloc((size_t)M * D * 2);
  bf16u* ctx = (bf16u*)alloc((size_t)M * D * 2);

  auto cvt = [&](const float* s, bf16u* d, int n) {
    int n4 = n / 4;
    mha_cvt4_kernel<<<(n4 + 255) / 256, 256, 0, stream>>>(s, d, n4);
  };
  cvt(x,  xb,  M * D);
  cvt(wq, wqb, D * D);
  cvt(wk, wkb, D * D);
  cvt(wv, wvb, D * D);
  cvt(wo, wob, D * D);

  dim3 ggrid(D / 128, M / 128);
  mha_gemm_bf16_kernel<true><<<ggrid, 256, 0, stream>>>(xb, wqb, (void*)Qb, M, D, D);
  mha_gemm_bf16_kernel<true><<<ggrid, 256, 0, stream>>>(xb, wkb, (void*)Kb, M, D, D);
  mha_gemm_bf16_kernel<true><<<ggrid, 256, 0, stream>>>(xb, wvb, (void*)Vb, M, D, D);

  mha_attn_kernel<<<dim3(S / 64, H, B), 128, 0, stream>>>(Qb, Kb, Vb, ctx, S, D, H);

  mha_gemm_bf16_kernel<false><<<ggrid, 256, 0, stream>>>(ctx, wob, d_out, M, D, D);
}
